// LinearTransformer_18365280157790
// MI455X (gfx1250) — matrix operand order repaired and checked by reading, not compiled, not run on hardware
//
#include <hip/hip_runtime.h>

// ---------------------------------------------------------------------------
// CDNA5 (gfx1250) bf16 WMMA implementation of the linear-attention transformer
// GEMM tiles fed by GLOBAL_LOAD_ASYNC_TO_LDS (ASYNCcnt-tracked, no VGPR staging)
// ---------------------------------------------------------------------------

typedef unsigned short u16;
typedef __attribute__((ext_vector_type(16))) __bf16 v16bf;
typedef __attribute__((ext_vector_type(8)))  float  v8f;
typedef __attribute__((ext_vector_type(4)))  int    v4i;

union FragAB { v16bf v; uint4 q[2]; };

#if __has_builtin(__builtin_amdgcn_global_load_async_to_lds_b128)
#define USE_ASYNC_LDS 1
#else
#define USE_ASYNC_LDS 0
#endif

__device__ __forceinline__ void g2l_b128(const void* g, void* l) {
#if USE_ASYNC_LDS
    __builtin_amdgcn_global_load_async_to_lds_b128(
        (__attribute__((address_space(1))) v4i*)(void*)g,
        (__attribute__((address_space(3))) v4i*)l, 0, 0);
#else
    (void)g; (void)l;
#endif
}

__device__ __forceinline__ void wait_async0() {
#if USE_ASYNC_LDS
#if __has_builtin(__builtin_amdgcn_s_wait_asynccnt)
    __builtin_amdgcn_s_wait_asynccnt(0);
#else
    asm volatile("s_wait_asynccnt 0" ::: "memory");
#endif
#endif
}

__device__ __forceinline__ u16 f2b(float x) {
    unsigned int u = __float_as_uint(x);
    unsigned int r = (u + 0x7FFFu + ((u >> 16) & 1u)) >> 16;   // RNE fp32 -> bf16
    return (u16)r;
}

__device__ __forceinline__ float gelu_t(float x) {
    const float c = 0.7978845608028654f;
    float t = tanhf(c * (x + 0.044715f * x * x * x));
    return 0.5f * x * (1.0f + t);
}

__device__ __forceinline__ float elu_p1(float x) {            // elu(x)+1
    return x > 0.0f ? x + 1.0f : __expf(x);
}

// ------------------------------ small kernels ------------------------------

// Build bf16 [B*S, T] input: rows s<L copy x, row s==L is the sinusoidal token.
__global__ void k_build_xfull(const float* __restrict__ x,
                              const float* __restrict__ nv,
                              u16* __restrict__ xf,
                              int S, int Ltok, int T) {
    int r = blockIdx.x;
    int b = r / S, s = r % S;
    u16* o = xf + (size_t)r * T;
    if (s < Ltok) {
        const float* xr = x + ((size_t)b * Ltok + s) * T;
        for (int i = threadIdx.x; i < T; i += blockDim.x) o[i] = f2b(xr[i]);
    } else {
        float v = nv[b];
        int h = T >> 1;
        for (int i = threadIdx.x; i < T; i += blockDim.x) {
            int t = (i < h) ? i : i - h;
            float lf = logf(1000.0f) * (float)t / (float)(h - 1);
            float ang = 6.283185307179586f * __expf(lf);
            float val = (i < h) ? __sinf(ang * v) : __cosf(ang * v);
            o[i] = f2b(val);
        }
    }
}

// fp32 [K,N] -> bf16 [N,K] (transpose + convert), 32x32 tiles via LDS.
__global__ void k_transpose_convert(const float* __restrict__ src,
                                    u16* __restrict__ dst, int K, int N) {
    __shared__ float t[32][33];
    int kb = blockIdx.y * 32, nb = blockIdx.x * 32;
    int tx = threadIdx.x, ty = threadIdx.y;            // 32 x 8
    #pragma unroll
    for (int i = 0; i < 32; i += 8)
        t[ty + i][tx] = src[(size_t)(kb + ty + i) * N + (nb + tx)];
    __syncthreads();
    #pragma unroll
    for (int i = 0; i < 32; i += 8)
        dst[(size_t)(nb + ty + i) * K + (kb + tx)] = f2b(t[tx][ty + i]);
}

__global__ void k_zero(float* __restrict__ p, long n) {
    long i = (long)blockIdx.x * blockDim.x + threadIdx.x;
    if (i < n) p[i] = 0.0f;
}

// KV[b,a] = sum_s K*V ; Ksum[b,a] = sum_s K  (chunked + atomic f32 adds)
__global__ void k_colreduce(const float* __restrict__ Kf,
                            const float* __restrict__ Vf,
                            float* __restrict__ kv, float* __restrict__ ks,
                            int S, int A, int nchunk) {
    int a  = blockIdx.x * blockDim.x + threadIdx.x;
    int b  = blockIdx.y;
    int ch = blockIdx.z;
    int rows = (S + nchunk - 1) / nchunk;
    int s0 = ch * rows;
    int s1 = min(S, s0 + rows);
    if (s0 >= s1) return;
    const float* kp = Kf + ((size_t)b * S + s0) * A + a;
    const float* vp = Vf + ((size_t)b * S + s0) * A + a;
    float akv = 0.0f, ak = 0.0f;
    for (int s = s0; s < s1; ++s) {
        float kk = *kp, vv = *vp;
        akv += kk * vv; ak += kk;
        kp += A; vp += A;
    }
    atomicAdd(&kv[b * A + a], akv);
    atomicAdd(&ks[b * A + a], ak);
}

// att = Q*KV / (Q*Ksum + 1e-6) -> bf16
__global__ void k_att(const float* __restrict__ Q,
                      const float* __restrict__ kv, const float* __restrict__ ks,
                      u16* __restrict__ att, int S, int A, long total) {
    long idx = (long)blockIdx.x * blockDim.x + threadIdx.x;
    if (idx >= total) return;
    int a = (int)(idx % A);
    long r = idx / A;
    int b = (int)(r / S);
    float q = Q[idx];
    float v = q * kv[b * A + a] / (q * ks[b * A + a] + 1e-6f);
    att[idx] = f2b(v);
}

// Row LayerNorm: ho/hb = LN(x)*sc + bi, one 256-thread block per row (E=1024)
__global__ void k_ln(const float* __restrict__ x,
                     const float* __restrict__ sc, const float* __restrict__ bi,
                     float* __restrict__ ho, u16* __restrict__ hb, int E) {
    __shared__ float sh[256], sh2[256];
    int r = blockIdx.x, tid = threadIdx.x;
    const float* xr = x + (size_t)r * E;
    float s = 0.0f, s2 = 0.0f;
    for (int i = tid; i < E; i += 256) { float v = xr[i]; s += v; s2 += v * v; }
    sh[tid] = s; sh2[tid] = s2;
    __syncthreads();
    for (int off = 128; off > 0; off >>= 1) {
        if (tid < off) { sh[tid] += sh[tid + off]; sh2[tid] += sh2[tid + off]; }
        __syncthreads();
    }
    float mean = sh[0] / (float)E;
    float var  = sh2[0] / (float)E - mean * mean;
    float inv  = rsqrtf(var + 1e-6f);
    for (int i = tid; i < E; i += 256) {
        float y = (xr[i] - mean) * inv * sc[i] + bi[i];
        ho[(size_t)r * E + i] = y;
        hb[(size_t)r * E + i] = f2b(y);
    }
}

// ------------------------------ WMMA GEMM ----------------------------------
// C[M,N] = epilogue(A[M,K](bf16) x Bt[N,K](bf16)^T + bias)
// 128x128 C-tile / workgroup, 8 waves of 32, each wave 32x64 via 2x4 WMMA.
// Tiles are DMA'd global->LDS with GLOBAL_LOAD_ASYNC_TO_LDS_B128 (ASYNCcnt).

enum { EPI_STORE = 0, EPI_EMBED = 1, EPI_ELU1 = 2, EPI_GELU_B16 = 3,
       EPI_GELU_RES = 4, EPI_OUT = 5 };

#define BM  128
#define BN  128
#define BK  64
#define BKP 72   // padded LDS row (elements): 144B rows, 16B-aligned frag loads

template<int EPI>
__global__ __launch_bounds__(256, 2)
void k_gemm(const u16* __restrict__ A, const u16* __restrict__ Bt,
            const float* __restrict__ bias, const float* __restrict__ res,
            const float* __restrict__ pe,
            float* __restrict__ outf, u16* __restrict__ outb,
            int Mrows, int N, int K, int S, int Ltok) {
    __shared__ u16 sA[2][BM * BKP];
    __shared__ u16 sB[2][BN * BKP];

    const int tid  = threadIdx.x;
    const int lane = tid & 31;
    const int wid  = tid >> 5;
    const int wm   = wid & 3;        // 4 waves along M (32 rows each)
    const int wn   = wid >> 2;       // 2 waves along N (64 cols each)
    const int m0   = blockIdx.x * BM;
    const int n0   = blockIdx.y * BN;

    v8f acc[2][4];
    v8f z = {0.f, 0.f, 0.f, 0.f, 0.f, 0.f, 0.f, 0.f};
    #pragma unroll
    for (int i = 0; i < 2; ++i)
        #pragma unroll
        for (int j = 0; j < 4; ++j) acc[i][j] = z;

    const int nkt = K / BK;

    // Copy one (BM x BK) A-tile and (BN x BK) B-tile into LDS buffer `buf`.
    // Async path: no VGPR staging; out-of-range A rows are skipped (their
    // stale LDS rows only feed C rows that the epilogue never stores).
    auto copyTile = [&](int kt, int buf) {
        #pragma unroll
        for (int i = 0; i < 4; ++i) {
            int c = tid + i * 256;              // 1024 uint4 chunks / tile
            int row = c >> 3, col = c & 7;
            int gr = m0 + row;
            const u16* gA = A + (size_t)gr * K + kt * BK + col * 8;
            const u16* gB = Bt + (size_t)(n0 + row) * K + kt * BK + col * 8;
            u16* lA = &sA[buf][row * BKP + col * 8];
            u16* lB = &sB[buf][row * BKP + col * 8];
#if USE_ASYNC_LDS
            if (gr < Mrows) g2l_b128(gA, lA);
            g2l_b128(gB, lB);
#else
            uint4 va = (gr < Mrows) ? *(const uint4*)gA : make_uint4(0u, 0u, 0u, 0u);
            *(uint4*)lA = va;
            *(uint4*)lB = *(const uint4*)gB;
#endif
        }
    };

    copyTile(0, 0);
    wait_async0();
    __syncthreads();

    for (int kt = 0; kt < nkt; ++kt) {
        int buf = kt & 1;
        bool more = (kt + 1 < nkt);
        if (more) copyTile(kt + 1, buf ^ 1);   // DMA next tiles while computing

        #pragma unroll
        for (int kk = 0; kk < BK; kk += 32) {
            FragAB fa[2], fb[4];
            // A fragment: 16x32 bf16. lanes0-15: M=lane, K=0..7 & 16..23;
            // lanes16-31: M=lane-16, K=8..15 & 24..31.
            #pragma unroll
            for (int mt = 0; mt < 2; ++mt) {
                int arow = wm * 32 + mt * 16 + (lane & 15);
                int kh = (lane >> 4) * 8;
                const u16* p = &sA[buf][arow * BKP + kk + kh];
                fa[mt].q[0] = *(const uint4*)(p);
                fa[mt].q[1] = *(const uint4*)(p + 16);
            }
            // B fragment: 32x16 bf16. lanes0-15: N=lane, K=0..7 & 16..23;
            // lanes16-31: N=lane-16, K=8..15 & 24..31.  (Bt stored [N][K])
            #pragma unroll
            for (int nt = 0; nt < 4; ++nt) {
                int brow = wn * 64 + nt * 16 + (lane & 15);
                int kq = (lane >> 4) * 8;
                const u16* p = &sB[buf][brow * BKP + kk + kq];
                fb[nt].q[0] = *(const uint4*)(p);
                fb[nt].q[1] = *(const uint4*)(p + 16);
            }
            #pragma unroll
            for (int mt = 0; mt < 2; ++mt)
                #pragma unroll
                for (int nt = 0; nt < 4; ++nt)
                    acc[mt][nt] = __builtin_amdgcn_wmma_f32_16x16x32_bf16(
                        false, fa[mt].v, false, fb[nt].v,
                        (short)0, acc[mt][nt], false, false);
        }

        if (more) {
            wait_async0();      // next tiles fully landed in LDS
            __syncthreads();    // visible to all waves; prev buffer free
        }
    }

    // Epilogue.  C/D layout: lanes0-15 hold M=vgpr, lanes16-31 hold M=vgpr+8; N=lane&15.
    #pragma unroll
    for (int mt = 0; mt < 2; ++mt) {
        #pragma unroll
        for (int nt = 0; nt < 4; ++nt) {
            #pragma unroll
            for (int v = 0; v < 8; ++v) {
                int m = (lane >> 4) * 8 + v;
                int n = lane & 15;
                int r = m0 + wm * 32 + mt * 16 + m;
                int c = n0 + wn * 64 + nt * 16 + n;
                if (r >= Mrows) continue;
                float val = acc[mt][nt][v] + bias[c];
                size_t o = (size_t)r * N + c;
                if (EPI == EPI_EMBED) {
                    val += pe[(size_t)(r % S) * N + c];
                    outf[o] = val; outb[o] = f2b(val);
                } else if (EPI == EPI_ELU1) {
                    outf[o] = elu_p1(val);
                } else if (EPI == EPI_STORE) {
                    outf[o] = val;
                } else if (EPI == EPI_GELU_B16) {
                    outb[o] = f2b(gelu_t(val));
                } else if (EPI == EPI_GELU_RES) {
                    outf[o] = gelu_t(val) + res[o];
                } else { // EPI_OUT : drop the noise token (s == L)
                    int s = r % S, b = r / S;
                    if (s < Ltok) outf[((size_t)b * Ltok + s) * N + c] = val;
                }
            }
        }
    }
}

// ------------------------------- host side ---------------------------------

extern "C" void kernel_launch(void* const* d_in, const int* in_sizes, int n_in,
                              void* d_out, int out_size, void* d_ws, size_t ws_size,
                              hipStream_t stream) {
    (void)in_sizes; (void)n_in; (void)out_size; (void)ws_size;

    const int Bb = 4, Ltok = 4096, T = 512, E = 1024, Af = 1024, F = 4096, NBl = 8;
    const int S = Ltok + 1;                 // 4097 tokens
    const long Nrows = (long)Bb * S;        // 16388 rows

    const float* x    = (const float*)d_in[0];
    const float* nv   = (const float*)d_in[1];
    const float* eW   = (const float*)d_in[2];
    const float* eB   = (const float*)d_in[3];
    const float* pe   = (const float*)d_in[4];
    const float* Wq   = (const float*)d_in[5];
    const float* bq   = (const float*)d_in[6];
    const float* Wk   = (const float*)d_in[7];
    const float* bk   = (const float*)d_in[8];
    const float* Wv   = (const float*)d_in[9];
    const float* bv   = (const float*)d_in[10];
    const float* Wo   = (const float*)d_in[11];
    const float* bo   = (const float*)d_in[12];
    const float* l1s  = (const float*)d_in[13];
    const float* l1b  = (const float*)d_in[14];
    const float* Wf1  = (const float*)d_in[15];
    const float* bf1  = (const float*)d_in[16];
    const float* Wf2  = (const float*)d_in[17];
    const float* bf2  = (const float*)d_in[18];
    const float* l2s  = (const float*)d_in[19];
    const float* l2b  = (const float*)d_in[20];
    const float* oW   = (const float*)d_in[21];
    const float* oB   = (const float*)d_in[22];
    float* out        = (float*)d_out;

    // workspace layout (256B aligned slabs)
    char* ws = (char*)d_ws;
    size_t off = 0;
    auto slab = [&](size_t bytes) -> char* {
        off = (off + 255) & ~(size_t)255;
        char* p = ws + off;
        off += bytes;
        return p;
    };
    u16*   xfull = (u16*)  slab((size_t)Nrows * T * 2);
    float* hf    = (float*)slab((size_t)Nrows * E * 4);
    u16*   hb    = (u16*)  slab((size_t)Nrows * E * 2);
    float* qf    = (float*)slab((size_t)Nrows * Af * 4);
    float* kf    = (float*)slab((size_t)Nrows * Af * 4);
    float* vf    = (float*)slab((size_t)Nrows * Af * 4);
    u16*   attb  = (u16*)  slab((size_t)Nrows * Af * 2);
    u16*   fb16  = (u16*)  slab((size_t)Nrows * F * 2);
    float* tmpf  = (float*)slab((size_t)Nrows * E * 4);
    float* kvb   = (float*)slab((size_t)2 * Bb * Af * 4);  // kv | ksum adjacent
    float* ksb   = kvb + (size_t)Bb * Af;
    u16*   wt    = (u16*)  slab((size_t)E * F * 2);        // transposed weight buffer

    const dim3 blk256(256), blkT(32, 8);
    const int Mt = (int)((Nrows + BM - 1) / BM);           // 129

    auto transpose = [&](const float* src, int Kd, int Nd) {
        k_transpose_convert<<<dim3(Nd / 32, Kd / 32), blkT, 0, stream>>>(src, wt, Kd, Nd);
    };

    // 1) build bf16 input incl. sinusoidal noise token
    k_build_xfull<<<dim3((unsigned)Nrows), dim3(128), 0, stream>>>(x, nv, xfull, S, Ltok, T);

    // 2) h = xfull @ embed_W + embed_b + pos_emb   (writes fp32 + bf16)
    transpose(eW, T, E);
    k_gemm<EPI_EMBED><<<dim3(Mt, E / BN), blk256, 0, stream>>>(
        xfull, wt, eB, nullptr, pe, hf, hb, (int)Nrows, E, T, S, Ltok);

    for (int b = 0; b < NBl; ++b) {
        const float* wq = Wq + (size_t)b * E * Af;  const float* bqb = bq + (size_t)b * Af;
        const float* wk = Wk + (size_t)b * E * Af;  const float* bkb = bk + (size_t)b * Af;
        const float* wv = Wv + (size_t)b * E * Af;  const float* bvb = bv + (size_t)b * Af;
        const float* wo = Wo + (size_t)b * Af * E;  const float* bob = bo + (size_t)b * E;
        const float* w1 = Wf1 + (size_t)b * E * F;  const float* b1b = bf1 + (size_t)b * F;
        const float* w2 = Wf2 + (size_t)b * F * E;  const float* b2b = bf2 + (size_t)b * E;

        // Q = elu(h@Wq+bq)+1 ; K = elu(h@Wk+bk)+1 ; V = h@Wv+bv
        transpose(wq, E, Af);
        k_gemm<EPI_ELU1><<<dim3(Mt, Af / BN), blk256, 0, stream>>>(
            hb, wt, bqb, nullptr, nullptr, qf, nullptr, (int)Nrows, Af, E, S, Ltok);
        transpose(wk, E, Af);
        k_gemm<EPI_ELU1><<<dim3(Mt, Af / BN), blk256, 0, stream>>>(
            hb, wt, bkb, nullptr, nullptr, kf, nullptr, (int)Nrows, Af, E, S, Ltok);
        transpose(wv, E, Af);
        k_gemm<EPI_STORE><<<dim3(Mt, Af / BN), blk256, 0, stream>>>(
            hb, wt, bvb, nullptr, nullptr, vf, nullptr, (int)Nrows, Af, E, S, Ltok);

        // KV and Ksum column reductions
        long nkv = (long)2 * Bb * Af;
        k_zero<<<dim3((unsigned)((nkv + 255) / 256)), blk256, 0, stream>>>(kvb, nkv);
        k_colreduce<<<dim3(Af / 256, Bb, 64), blk256, 0, stream>>>(kf, vf, kvb, ksb, S, Af, 64);

        // att = Q * KV / (Q * Ksum + 1e-6)  -> bf16
        long tot = Nrows * Af;
        k_att<<<dim3((unsigned)((tot + 255) / 256)), blk256, 0, stream>>>(
            qf, kvb, ksb, attb, S, Af, tot);

        // a = gelu(att@Wo+bo) + h ; h = LN1(a)
        transpose(wo, Af, E);
        k_gemm<EPI_GELU_RES><<<dim3(Mt, E / BN), blk256, 0, stream>>>(
            attb, wt, bob, hf, nullptr, tmpf, nullptr, (int)Nrows, E, Af, S, Ltok);
        k_ln<<<dim3((unsigned)Nrows), blk256, 0, stream>>>(tmpf, l1s + (size_t)b * E,
                                                          l1b + (size_t)b * E, hf, hb, E);

        // f = gelu(h@Wf1+bf1) (bf16) ; t = gelu(f@Wf2+bf2) + h ; h = LN2(t)
        transpose(w1, E, F);
        k_gemm<EPI_GELU_B16><<<dim3(Mt, F / BN), blk256, 0, stream>>>(
            hb, wt, b1b, nullptr, nullptr, nullptr, fb16, (int)Nrows, F, E, S, Ltok);
        transpose(w2, F, E);
        k_gemm<EPI_GELU_RES><<<dim3(Mt, E / BN), blk256, 0, stream>>>(
            fb16, wt, b2b, hf, nullptr, tmpf, nullptr, (int)Nrows, E, F, S, Ltok);
        k_ln<<<dim3((unsigned)Nrows), blk256, 0, stream>>>(tmpf, l2s + (size_t)b * E,
                                                          l2b + (size_t)b * E, hf, hb, E);
    }

    // out = h @ out_W + out_b, dropping the noise token
    transpose(oW, E, T);
    k_gemm<EPI_OUT><<<dim3(Mt, T / BN), blk256, 0, stream>>>(
        hb, wt, oB, nullptr, nullptr, out, nullptr, (int)Nrows, T, E, S, Ltok);
}
